// DeepseekV4HashRouter_73813307949707
// MI455X (gfx1250) — matrix operand order repaired and checked by reading, not compiled, not run on hardware
//
#include <hip/hip_runtime.h>

// DeepSeek-V4 hash router, fused on gfx1250 (MI455X).
// GEMM via bf16x3-split V_WMMA_F32_16X16X32_BF16 (fp32-class accuracy on the
// bf16 tensor pipe). Weight hi/lo split precomputed once into d_ws (loop
// invariant, 2 MB). Hidden tile staged into LDS with the CDNA5 async
// global->LDS path (ASYNCcnt). Then sqrt(softplus), hash gather, renormalize,
// dense scatter — all workgroup-local.

typedef __attribute__((ext_vector_type(16))) __bf16 v16bf;
typedef __attribute__((ext_vector_type(8)))  float  v8f;

#define N_TOK 16384   // B*S
#define DIM   2048    // hidden size
#define NE    256     // experts
#define TOPK  8
#define ROWS  32      // token rows per workgroup
#define KC    32      // K chunk (one bf16 WMMA depth)

// ---------------------------------------------------------------------------
// Prep: split fp32 weight into hi/lo bf16 planes (runs once per launch; weight
// is 2 MB so this is noise next to the 134 MB hidden stream).
// ---------------------------------------------------------------------------
__global__ __launch_bounds__(256)
void weight_split_kernel(const float* __restrict__ w,
                         __bf16* __restrict__ w_hi,
                         __bf16* __restrict__ w_lo)
{
    int i = blockIdx.x * 256 + threadIdx.x;        // grid exactly covers E*D
    float  x = w[i];
    __bf16 h = (__bf16)x;
    w_hi[i] = h;
    w_lo[i] = (__bf16)(x - (float)h);
}

// ---------------------------------------------------------------------------
// Fused router kernel. PRESPLIT: B fragments come pre-split from d_ws.
// ---------------------------------------------------------------------------
template <bool PRESPLIT>
__global__ __launch_bounds__(512)
void DeepseekV4HashRouter_kernel(const float*  __restrict__ hidden,
                                 const int*    __restrict__ token_ids,
                                 const float*  __restrict__ weight,
                                 const __bf16* __restrict__ w_hi,
                                 const __bf16* __restrict__ w_lo,
                                 const int*    __restrict__ tid2eid,
                                 float*        __restrict__ out)
{
    __shared__ float lds_a[ROWS][KC + 4];     // staged hidden tile (fp32)
    __shared__ float lds_s[ROWS][NE + 4];     // full score rows
    __shared__ int   lds_ix[ROWS][TOPK];
    __shared__ float lds_wt[ROWS][TOPK];

    const int tid     = threadIdx.x;
    const int wave    = tid >> 5;             // 0..15 -> expert column tile
    const int lane    = tid & 31;
    const int rowBase = blockIdx.x * ROWS;
    const int colBase = wave * 16;

    // WMMA fragment geometry (wave32; A and B both use contiguous k = 16*(lane>>4) + s)
    const int m   = lane & 15;                // A row / C col index
    const int kbA = (lane >> 4) * 16;         // A: one 16-long contiguous K run
    const int kbB = (lane >> 4) * 16;         // B: one 16-long contiguous K run
    const int ecol = colBase + m;             // this lane's expert column

    const float*  wrow   = weight + (size_t)ecol * DIM + kbB;
    const __bf16* wrowhi = PRESPLIT ? (w_hi + (size_t)ecol * DIM + kbB) : nullptr;
    const __bf16* wrowlo = PRESPLIT ? (w_lo + (size_t)ecol * DIM + kbB) : nullptr;

    // Async global->LDS staging: each thread moves 2 floats (8 bytes).
    const int      si       = tid * 2;                 // 0..1022, even
    const int      sr       = si >> 5, sc = si & 31;
    const unsigned lds_dst  = (unsigned)(uintptr_t)&lds_a[sr][sc];  // WG-relative LDS addr
    const float*   gsrc_row = hidden + (size_t)(rowBase + sr) * DIM + sc;

    v8f acc0 = {}; v8f acc1 = {};

    for (int k0 = 0; k0 < DIM; k0 += KC) {
        // ---- stage hidden tile (32x32 f32) straight into LDS (ASYNCcnt) ----
        {
            const float* gsrc = gsrc_row + k0;
            asm volatile("global_load_async_to_lds_b64 %0, %1, off"
                         :: "v"(lds_dst), "v"(gsrc) : "memory");
        }

        // ---- B fragments: 16 contiguous K values of this lane's expert ----
        v16bf b_hi, b_lo;
        if (PRESPLIT) {
            b_hi = *(const v16bf*)(wrowhi + k0);       // 2x b128 loads
            b_lo = *(const v16bf*)(wrowlo + k0);
        } else {
            const float* p = wrow + k0;
#pragma unroll
            for (int j = 0; j < 16; ++j) {
                float  x = p[j];
                __bf16 h = (__bf16)x;
                b_hi[j] = h;
                b_lo[j] = (__bf16)(x - (float)h);
            }
        }

        // ---- wait for this wave's async copy, then workgroup barrier ----
        asm volatile("s_wait_asynccnt 0x0" ::: "memory");
        __syncthreads();

        // ---- A fragments for two 16-row tiles, split hi/lo bf16 ----
        v16bf a_hi0, a_lo0, a_hi1, a_lo1;
#pragma unroll
        for (int j = 0; j < 8; ++j) {
            float x0 = lds_a[m][kbA + j];
            float x1 = lds_a[m][kbA + 8 + j];
            __bf16 h0 = (__bf16)x0, h1 = (__bf16)x1;
            a_hi0[j]     = h0; a_lo0[j]     = (__bf16)(x0 - (float)h0);
            a_hi0[8 + j] = h1; a_lo0[8 + j] = (__bf16)(x1 - (float)h1);

            float y0 = lds_a[16 + m][kbA + j];
            float y1 = lds_a[16 + m][kbA + 8 + j];
            __bf16 g0 = (__bf16)y0, g1 = (__bf16)y1;
            a_hi1[j]     = g0; a_lo1[j]     = (__bf16)(y0 - (float)g0);
            a_hi1[8 + j] = g1; a_lo1[8 + j] = (__bf16)(y1 - (float)g1);
        }

        // ---- bf16x3 accumulation: hi*hi + lo*hi + hi*lo ----
        acc0 = __builtin_amdgcn_wmma_f32_16x16x32_bf16(false, a_hi0, false, b_hi, (short)0, acc0, false, false);
        acc0 = __builtin_amdgcn_wmma_f32_16x16x32_bf16(false, a_lo0, false, b_hi, (short)0, acc0, false, false);
        acc0 = __builtin_amdgcn_wmma_f32_16x16x32_bf16(false, a_hi0, false, b_lo, (short)0, acc0, false, false);
        acc1 = __builtin_amdgcn_wmma_f32_16x16x32_bf16(false, a_hi1, false, b_hi, (short)0, acc1, false, false);
        acc1 = __builtin_amdgcn_wmma_f32_16x16x32_bf16(false, a_lo1, false, b_hi, (short)0, acc1, false, false);
        acc1 = __builtin_amdgcn_wmma_f32_16x16x32_bf16(false, a_hi1, false, b_lo, (short)0, acc1, false, false);

        __syncthreads();   // protect lds_a before next chunk's async stores
    }

    // ---- scores = sqrt(softplus(logit)) -> LDS (C: elem r -> row r+8*(lane>=16)) ----
    {
        const int mofs = (lane >> 4) * 8;
#pragma unroll
        for (int r = 0; r < 8; ++r) {
            float x0 = acc0[r];
            float s0 = sqrtf(fmaxf(x0, 0.0f) + __logf(1.0f + __expf(-fabsf(x0))));
            lds_s[mofs + r][colBase + m] = s0;
            float x1 = acc1[r];
            float s1 = sqrtf(fmaxf(x1, 0.0f) + __logf(1.0f + __expf(-fabsf(x1))));
            lds_s[16 + mofs + r][colBase + m] = s1;
        }
    }
    __syncthreads();

    // ---- hash gather + renormalize (one thread per token row) ----
    if (tid < ROWS) {
        const int n  = rowBase + tid;
        const int tk = token_ids[n];
        int   ix[TOPK];
        float sv[TOPK];
        float sum = 0.0f;
#pragma unroll
        for (int j = 0; j < TOPK; ++j) {
            int e = tid2eid[(size_t)tk * TOPK + j];
            ix[j] = e;
            float v = lds_s[tid][e];
            sv[j] = v;
            sum += v;
        }
        const float scale = 2.5f / fmaxf(sum, 1e-12f);
#pragma unroll
        for (int j = 0; j < TOPK; ++j) {
            lds_ix[tid][j] = ix[j];
            lds_wt[tid][j] = sv[j] * scale;
        }
    }
    __syncthreads();

    // ---- dense scatter: probs[N,E] then routing_map[N,E] (as 0/1 floats) ----
    {
        const int row = tid >> 4;     // 0..31
        const int g   = tid & 15;
        const int n   = rowBase + row;
        int   ix[TOPK];
        float wv[TOPK];
#pragma unroll
        for (int j = 0; j < TOPK; ++j) { ix[j] = lds_ix[row][j]; wv[j] = lds_wt[row][j]; }

        float* probs = out + (size_t)n * NE;
        float* rmap  = out + (size_t)N_TOK * NE + (size_t)n * NE;
#pragma unroll
        for (int j2 = 0; j2 < 16; ++j2) {
            int e = j2 * 16 + g;      // 16 threads of a row write consecutive experts
            float val = 0.0f, flag = 0.0f;
#pragma unroll
            for (int j = 0; j < TOPK; ++j) {
                if (ix[j] == e) { val = wv[j]; flag = 1.0f; }
            }
            probs[e] = val;
            rmap[e]  = flag;
        }
    }
}

extern "C" void kernel_launch(void* const* d_in, const int* in_sizes, int n_in,
                              void* d_out, int out_size, void* d_ws, size_t ws_size,
                              hipStream_t stream) {
    (void)in_sizes; (void)n_in; (void)out_size;
    const float* hidden    = (const float*)d_in[0];
    const int*   token_ids = (const int*)  d_in[1];
    const float* weight    = (const float*)d_in[2];
    const int*   tid2eid   = (const int*)  d_in[3];
    float*       out       = (float*)d_out;

    dim3 grid(N_TOK / ROWS);   // 512 workgroups
    dim3 block(512);           // 16 wave32 waves

    const size_t split_bytes = (size_t)NE * DIM * sizeof(__bf16) * 2;  // 2 MB
    if (d_ws != nullptr && ws_size >= split_bytes) {
        __bf16* w_hi = (__bf16*)d_ws;
        __bf16* w_lo = w_hi + (size_t)NE * DIM;
        weight_split_kernel<<<(NE * DIM) / 256, 256, 0, stream>>>(weight, w_hi, w_lo);
        DeepseekV4HashRouter_kernel<true><<<grid, block, 0, stream>>>(
            hidden, token_ids, weight, w_hi, w_lo, tid2eid, out);
    } else {
        DeepseekV4HashRouter_kernel<false><<<grid, block, 0, stream>>>(
            hidden, token_ids, weight, nullptr, nullptr, tid2eid, out);
    }
}
